// MultiHeadAttention_56066503082144
// MI455X (gfx1250) — matrix operand order repaired and checked by reading, not compiled, not run on hardware
//
#include <hip/hip_runtime.h>
#include <math.h>

#define B_  8
#define S_  1024
#define D_  512
#define H_  8
#define DH_ 64
#define N_  (B_*S_)   // 8192 rows

typedef __attribute__((ext_vector_type(8)))  _Float16 v8h;
typedef __attribute__((ext_vector_type(16))) _Float16 v16h;
typedef __attribute__((ext_vector_type(8)))  float    v8f;
typedef __attribute__((ext_vector_type(4)))  unsigned int v4u;
typedef __attribute__((ext_vector_type(4)))  int      v4i;
typedef __attribute__((ext_vector_type(8)))  int      v8i;

#if defined(__AMDGCN__) && __has_builtin(__builtin_amdgcn_tensor_load_to_lds)
#define HAVE_TDM 1
#else
#define HAVE_TDM 0
#endif

// ---- WMMA fragment helpers (layouts per CDNA5 ISA 7.12.2, wave32) ----
static __device__ __forceinline__ v16h ld16h(const _Float16* p0, const _Float16* p1) {
  union { v16h v; v8h h[2]; } u;
  u.h[0] = *(const v8h*)p0;
  u.h[1] = *(const v8h*)p1;
  return u.v;
}
// A fragment 16x32 f16: lane holds row (lane&15); K chunks at +8g and +16+8g
static __device__ __forceinline__ v16h loadA(const _Float16* row, int g) {
  return ld16h(row + 8*g, row + 16 + 8*g);
}
// B fragment 32x16 f16: lane holds col (lane&15); K chunks at +8g and +16+8g
static __device__ __forceinline__ v16h loadB(const _Float16* col, int g) {
  return ld16h(col + 8*g, col + 16 + 8*g);
}
static __device__ __forceinline__ v8f wmma16(v16h a, v16h b, v8f c) {
  return __builtin_amdgcn_wmma_f32_16x16x32_f16(false, a, false, b, (short)0, c, false, false);
}
static __device__ __forceinline__ v8f vzero8() {
  v8f z = {0.f,0.f,0.f,0.f,0.f,0.f,0.f,0.f};
  return z;
}

// ---- 1) f32 -> f16 conversions ----
__global__ void cvt_x_kernel(const float* __restrict__ x, _Float16* __restrict__ xh, int n) {
  int i = blockIdx.x * blockDim.x + threadIdx.x;
  if (i < n) xh[i] = (_Float16)x[i];
}
// Transposed weights WT[m][n*512 + k] = W[m][k*512 + n] so B-fragment loads are contiguous
__global__ void cvt_w_kernel(const float* __restrict__ Wq, const float* __restrict__ Wk,
                             const float* __restrict__ Wv, _Float16* __restrict__ wt) {
  int i = blockIdx.x * blockDim.x + threadIdx.x;      // i = m*2^18 + n*512 + k
  if (i >= 3 * D_ * D_) return;
  int m = i >> 18;
  int rem = i & ((D_*D_) - 1);
  int n = rem >> 9, k = rem & (D_ - 1);
  const float* W = (m == 0) ? Wq : (m == 1) ? Wk : Wv;
  wt[i] = (_Float16)W[k * D_ + n];
}

// ---- 2) QKV projection: P = relu(X @ W + b), stored f16 in [b,h,s,dh] ----
// Each wave computes a 16x64 strip: A fragment reused across 4 B fragments.
__global__ void qkv_gemm_kernel(const _Float16* __restrict__ Xh,
                                const _Float16* __restrict__ WT,
                                const float* __restrict__ bq,
                                const float* __restrict__ bk,
                                const float* __restrict__ bv,
                                _Float16* __restrict__ Qh,
                                _Float16* __restrict__ Kh,
                                _Float16* __restrict__ Vh) {
  const int lane = threadIdx.x & 31;
  const int wv   = threadIdx.x >> 5;       // 4 waves per block
  const int g    = lane >> 4;
  const int ln   = lane & 15;
  const int rowt = blockIdx.x;             // 512 row tiles of 16
  const int colb = (blockIdx.y * 4 + wv) * 64;   // 16x64 strip per wave
  const int z    = blockIdx.z;             // 0:Q 1:K 2:V

  const _Float16* Wz  = WT + z * (D_ * D_);
  const float* bias   = (z == 0) ? bq : (z == 1) ? bk : bv;
  _Float16* dst       = (z == 0) ? Qh : (z == 1) ? Kh : Vh;

  const _Float16* arow = Xh + (rowt * 16 + ln) * D_;
  const _Float16* bc0 = Wz + (colb +  0 + ln) * D_;
  const _Float16* bc1 = Wz + (colb + 16 + ln) * D_;
  const _Float16* bc2 = Wz + (colb + 32 + ln) * D_;
  const _Float16* bc3 = Wz + (colb + 48 + ln) * D_;

  v8f acc[4] = {vzero8(), vzero8(), vzero8(), vzero8()};
  for (int kb = 0; kb < D_; kb += 32) {
    if (kb + 64 < D_) {
      __builtin_prefetch(arow + kb + 64, 0, 3);
      __builtin_prefetch(bc0 + kb + 64, 0, 3);
    }
    v16h a = loadA(arow + kb, g);
    acc[0] = wmma16(a, loadB(bc0 + kb, g), acc[0]);
    acc[1] = wmma16(a, loadB(bc1 + kb, g), acc[1]);
    acc[2] = wmma16(a, loadB(bc2 + kb, g), acc[2]);
    acc[3] = wmma16(a, loadB(bc3 + kb, g), acc[3]);
  }
  for (int j = 0; j < 4; ++j) {
    const int ncol = colb + j * 16 + ln;
    const float bz = bias[ncol];
    const int h = ncol >> 6, dd = ncol & 63;
    for (int r = 0; r < 8; ++r) {
      int row = rowt * 16 + r + 8 * g;     // D layout: m = r + 8g
      int bb = row >> 10, s = row & (S_ - 1);
      float v = acc[j][r] + bz;
      v = v > 0.f ? v : 0.f;
      dst[(((bb * H_ + h) * S_) + s) * DH_ + dd] = (_Float16)v;
    }
  }
}

// ---- 3) fused attention per (b,h): K,V^T resident in LDS; query-axis softmax ----
__global__ void attn_kernel(const _Float16* __restrict__ Qh,
                            const _Float16* __restrict__ Kh,
                            const _Float16* __restrict__ Vh,
                            const float* __restrict__ x,
                            float* __restrict__ y) {
  extern __shared__ char smem[];
  _Float16* Ks   = (_Float16*)smem;                  // [1024][64]  128 KB (LDS offset 0)
  _Float16* Vt   = (_Float16*)(smem + 131072);       // [64][1024]  128 KB (transposed)
  _Float16* Psc  = (_Float16*)(smem + 262144);       // 8 waves x 16x32 = 8 KB
  float*    cmaxS = (float*)(smem + 270336);         // [1024] scaled column max
  float*    csumR = (float*)(smem + 274432);         // [1024] reciprocal column sum

  const int tid = threadIdx.x;
  const int lane = tid & 31, wv = tid >> 5;          // 8 waves
  const int g = lane >> 4, ln = lane & 15;
  const int bh = blockIdx.x;                         // b*8 + h
  const int b = bh >> 3, h = bh & 7;
  const _Float16* Qbh = Qh + bh * (S_ * DH_);
  const _Float16* Kbh = Kh + bh * (S_ * DH_);
  const _Float16* Vbh = Vh + bh * (S_ * DH_);
  const float sc = 0.125f;                           // 1/sqrt(64)

  // ---- stage 1: K -> LDS via TDM (first 64KB) + async global->LDS (rest);
  //      V -> LDS transposed via VALU path.
#if HAVE_TDM
  if (tid == 0) {
    // 2-D D# descriptor: one 8192-element row of 8-byte elements (64 KB) at LDS 0.
    unsigned long long ga = (unsigned long long)(uintptr_t)Kbh;
    v4u g0;
    g0[0] = 1u;                                        // count=1 (user descriptor)
    g0[1] = 0u;                                        // lds_addr = 0 (Ks base)
    g0[2] = (unsigned)(ga & 0xFFFFFFFFu);              // global_addr[31:0]
    g0[3] = (unsigned)((ga >> 32) & 0x01FFFFFFu)       // global_addr[56:32]
            | (2u << 30);                              // type = 2 ("image")
    v8i g1;
    g1[0] = (int)(3u << 16);                           // workgroup_mask=0, data_size=8B
    g1[1] = (int)((8192u & 0xFFFFu) << 16);            // tensor_dim0[15:0] @ bits63:48
    g1[2] = (int)((8192u >> 16) | (1u << 16));         // tensor_dim0[31:16] | tensor_dim1=1
    g1[3] = (int)(8192u << 16);                        // tile_dim0 = 8192
    g1[4] = 1;                                         // tile_dim1 = 1, tile_dim2 = 0
    g1[5] = 8192;                                      // tensor_dim0_stride[31:0]
    g1[6] = (int)((8192u & 0xFFFFu) << 16);            // dim0_stride[47:32]=0 | dim1_stride lo
    g1[7] = 0;
    v4i g2 = {0, 0, 0, 0};
    v4i g3 = {0, 0, 0, 0};
#if __clang_major__ >= 23
    v8i g4 = {0, 0, 0, 0, 0, 0, 0, 0};
    __builtin_amdgcn_tensor_load_to_lds(g0, g1, g2, g3, g4, 0);
#else
    __builtin_amdgcn_tensor_load_to_lds(g0, g1, g2, g3, 0);
#endif
    __builtin_amdgcn_s_wait_tensorcnt(0);
  }
  const int kchunk0 = 4096;   // first 4096 x 16B chunks handled by TDM
#else
  const int kchunk0 = 0;
#endif
  // remaining K bytes via async global->LDS loads (ASYNCcnt path)
  for (int i = kchunk0 + tid; i < 8192; i += 256) {
    unsigned ldsOff = (unsigned)(i * 16);                                  // Ks at LDS 0
    unsigned long long gp = (unsigned long long)(uintptr_t)Kbh
                            + (unsigned long long)i * 16ull;
    asm volatile("global_load_async_to_lds_b128 %0, %1, off"
                 :: "v"(ldsOff), "v"(gp) : "memory");
  }
  asm volatile("s_wait_asynccnt 0x0" ::: "memory");
  for (int i = tid; i < S_ * DH_; i += blockDim.x) {
    int k = i >> 6, d = i & 63;
    Vt[d * S_ + k] = Vbh[i];
  }
  __syncthreads();

  // ---- stage 2: column stats (softmax over the QUERY axis) — wave owns key tiles
  for (int kt = wv; kt < 64; kt += 8) {
    const _Float16* kcol = Ks + (kt * 16 + ln) * DH_;
    v16h b0 = loadB(kcol, g);          // dh 0..31
    v16h b1 = loadB(kcol + 32, g);     // dh 32..63
    float cmax = -3.0e38f;
    for (int qt = 0; qt < 64; ++qt) {
      const _Float16* qrow = Qbh + (qt * 16 + ln) * DH_;
      v8f s = vzero8();
      s = wmma16(loadA(qrow, g), b0, s);
      s = wmma16(loadA(qrow + 32, g), b1, s);
      for (int r = 0; r < 8; ++r) cmax = fmaxf(cmax, s[r]);
    }
    cmax = fmaxf(cmax, __shfl_xor(cmax, 16));
    float cm = cmax * sc;
    float csum = 0.f;
    for (int qt = 0; qt < 64; ++qt) {
      const _Float16* qrow = Qbh + (qt * 16 + ln) * DH_;
      v8f s = vzero8();
      s = wmma16(loadA(qrow, g), b0, s);
      s = wmma16(loadA(qrow + 32, g), b1, s);
      for (int r = 0; r < 8; ++r) csum += __expf(s[r] * sc - cm);
    }
    csum += __shfl_xor(csum, 16);
    if (lane < 16) { cmaxS[kt * 16 + ln] = cm; csumR[kt * 16 + ln] = csum; }
  }
  __syncthreads();
  for (int i = tid; i < S_; i += blockDim.x) csumR[i] = 1.f / csumR[i];
  __syncthreads();

  // ---- stage 3: out = P @ V (recompute scores, normalize, PV via WMMA) — wave owns q tiles
  _Float16* ps = Psc + wv * (16 * 32);
  for (int qt = wv; qt < 64; qt += 8) {
    const _Float16* qrow = Qbh + (qt * 16 + ln) * DH_;
    v16h qa0 = loadA(qrow, g);
    v16h qa1 = loadA(qrow + 32, g);
    v8f acc0 = vzero8(), acc1 = vzero8(), acc2 = vzero8(), acc3 = vzero8();
    for (int kb = 0; kb < S_; kb += 32) {
      const _Float16* k0 = Ks + (kb + ln) * DH_;
      const _Float16* k1 = Ks + (kb + 16 + ln) * DH_;
      v8f s0 = vzero8(), s1 = vzero8();
      s0 = wmma16(qa0, loadB(k0, g), s0);
      s0 = wmma16(qa1, loadB(k0 + 32, g), s0);
      s1 = wmma16(qa0, loadB(k1, g), s1);
      s1 = wmma16(qa1, loadB(k1 + 32, g), s1);
      int c0 = kb + ln, c1 = c0 + 16;
      float m0 = cmaxS[c0], z0 = csumR[c0];
      float m1 = cmaxS[c1], z1 = csumR[c1];
      for (int r = 0; r < 8; ++r) {
        int m = r + 8 * g;                     // D-layout row -> A-layout staging
        ps[m * 32 + ln]      = (_Float16)(__expf(s0[r] * sc - m0) * z0);
        ps[m * 32 + 16 + ln] = (_Float16)(__expf(s1[r] * sc - m1) * z1);
      }
      v16h pa = loadA(ps + ln * 32, g);        // same-wave DS ops are in-order
      acc0 = wmma16(pa, loadB(Vt + (0 * 16 + ln) * S_ + kb, g), acc0);
      acc1 = wmma16(pa, loadB(Vt + (1 * 16 + ln) * S_ + kb, g), acc1);
      acc2 = wmma16(pa, loadB(Vt + (2 * 16 + ln) * S_ + kb, g), acc2);
      acc3 = wmma16(pa, loadB(Vt + (3 * 16 + ln) * S_ + kb, g), acc3);
    }
    for (int r = 0; r < 8; ++r) {
      int row = b * S_ + qt * 16 + r + 8 * g;
      int base = row * D_ + h * DH_;
      y[base +  0 + ln] = acc0[r] + x[base +  0 + ln];
      y[base + 16 + ln] = acc1[r] + x[base + 16 + ln];
      y[base + 32 + ln] = acc2[r] + x[base + 32 + ln];
      y[base + 48 + ln] = acc3[r] + x[base + 48 + ln];
    }
  }
}

// ---- 4) BatchNorm stats per channel over (B,S), biased variance ----
__global__ void bn_stats_kernel(const float* __restrict__ y,
                                float* __restrict__ mean, float* __restrict__ invstd) {
  __shared__ float sh[256], sh2[256];
  int c = blockIdx.x, t = threadIdx.x;
  float s = 0.f, s2 = 0.f;
  for (int r = t; r < N_; r += 256) { float v = y[r * D_ + c]; s += v; s2 += v * v; }
  sh[t] = s; sh2[t] = s2; __syncthreads();
  for (int o = 128; o > 0; o >>= 1) {
    if (t < o) { sh[t] += sh[t + o]; sh2[t] += sh2[t + o]; }
    __syncthreads();
  }
  if (t == 0) {
    float m = sh[0] * (1.f / N_);
    float var = sh2[0] * (1.f / N_) - m * m;
    mean[c] = m;
    invstd[c] = rsqrtf(var + 1e-5f);
  }
}

// ---- 5) BatchNorm apply ----
__global__ void bn_apply_kernel(const float* __restrict__ y,
                                const float* __restrict__ mean,
                                const float* __restrict__ invstd,
                                const float* __restrict__ gamma,
                                const float* __restrict__ beta,
                                float* __restrict__ out) {
  int i = blockIdx.x * blockDim.x + threadIdx.x;
  if (i >= N_ * D_) return;
  int c = i & (D_ - 1);
  out[i] = (y[i] - mean[c]) * invstd[c] * gamma[c] + beta[c];
}

extern "C" void kernel_launch(void* const* d_in, const int* in_sizes, int n_in,
                              void* d_out, int out_size, void* d_ws, size_t ws_size,
                              hipStream_t stream) {
  (void)in_sizes; (void)n_in; (void)out_size; (void)ws_size;
  const float* x     = (const float*)d_in[0];
  const float* Wq    = (const float*)d_in[1];
  const float* bq    = (const float*)d_in[2];
  const float* Wk    = (const float*)d_in[3];
  const float* bk    = (const float*)d_in[4];
  const float* Wv    = (const float*)d_in[5];
  const float* bv    = (const float*)d_in[6];
  const float* gamma = (const float*)d_in[7];
  const float* beta  = (const float*)d_in[8];

  char* ws = (char*)d_ws;
  _Float16* Xh  = (_Float16*)(ws);                        //  8 MB
  _Float16* WT  = (_Float16*)(ws + 8388608);              //  1.5 MB
  _Float16* Qh  = (_Float16*)(ws + 9961472);              //  8 MB [b,h,s,dh]
  _Float16* Kh  = (_Float16*)(ws + 18350080);             //  8 MB
  _Float16* Vh  = (_Float16*)(ws + 26738688);             //  8 MB
  float*    Yb  = (float*)   (ws + 35127296);             // 16 MB  (pre-BN y)
  float*    Mn  = (float*)   (ws + 51904512);             // 2 KB
  float*    Iv  = (float*)   (ws + 51906560);             // 2 KB

  cvt_x_kernel<<<(N_ * D_ + 255) / 256, 256, 0, stream>>>(x, Xh, N_ * D_);
  cvt_w_kernel<<<(3 * D_ * D_ + 255) / 256, 256, 0, stream>>>(Wq, Wk, Wv, WT);

  qkv_gemm_kernel<<<dim3(N_ / 16, 2, 3), 128, 0, stream>>>(
      Xh, WT, bq, bk, bv, Qh, Kh, Vh);

  // 64 workgroups (one per b,h); 272 KB dynamic LDS (<= 320 KB WGP LDS)
  attn_kernel<<<B_ * H_, 256, 278528, stream>>>(Qh, Kh, Vh, x, Yb);

  bn_stats_kernel<<<D_, 256, 0, stream>>>(Yb, Mn, Iv);
  bn_apply_kernel<<<(N_ * D_ + 255) / 256, 256, 0, stream>>>(
      Yb, Mn, Iv, gamma, beta, (float*)d_out);
}
